// NoisyTopkRouter_57140244906729
// MI455X (gfx1250) — matrix operand order repaired and checked by reading, not compiled, not run on hardware
//
#include <hip/hip_runtime.h>
#include <hip/hip_bf16.h>
#include <math.h>

typedef __attribute__((ext_vector_type(16))) __bf16 v16bf;
typedef __attribute__((ext_vector_type(8)))  __bf16 v8bf;
typedef __attribute__((ext_vector_type(8)))  float  v8f;
typedef __attribute__((ext_vector_type(4)))  float  f32x4;

#define B_SZ   4
#define S_SZ   4096
#define D_SZ   2048
#define E_SZ   16
#define K_TOP  2
#define ROWS   (B_SZ * S_SZ)      // 16384
#define NTILES (ROWS / 16)        // 1024
// Padded LDS row stride (bf16 elems): 2056*2 B = 4112 B = 1028 dwords.
// 1028 mod 64 == 4 -> the 16 column-lanes map to distinct 4-bank groups,
// and 4112 is a multiple of 16 so ds_load_b128 alignment holds.
#define WSTRIDE 2056

__global__ __launch_bounds__(256) void noisy_topk_router_kernel(
    const float* __restrict__ x,        // (ROWS, D)
    const float* __restrict__ w_route,  // (D, E) row-major
    const float* __restrict__ b_route,  // (E)
    const float* __restrict__ w_noise,  // (D, E)
    const float* __restrict__ b_noise,  // (E)
    const float* __restrict__ eps,      // (ROWS, E)
    float*       __restrict__ out_probs,// (ROWS, E)
    int*         __restrict__ out_idx,  // (ROWS, 2)
    float*       __restrict__ aux_part) // (E) partial per-expert prob sums
{
  extern __shared__ __align__(16) __bf16 smem[];
  __bf16* wt_r = smem;                    // [E_SZ][WSTRIDE] transposed bf16 route
  __bf16* wt_n = smem + E_SZ * WSTRIDE;   // [E_SZ][WSTRIDE] transposed bf16 noise

  // ---- Stage both weight matrices into LDS: transpose + f32->bf16, once per block.
  // idx = kp*16 + e  -> consecutive threads read consecutive experts (coalesced),
  // and each thread packs two consecutive-K bf16 into one b32 LDS store.
  for (int idx = threadIdx.x; idx < (D_SZ / 2) * E_SZ; idx += blockDim.x) {
    const int e = idx & 15;
    const int k = (idx >> 4) * 2;
    union { __bf16 h[2]; unsigned u; } pr, pn;
    pr.h[0] = (__bf16)w_route[(size_t)k * E_SZ + e];
    pr.h[1] = (__bf16)w_route[(size_t)(k + 1) * E_SZ + e];
    pn.h[0] = (__bf16)w_noise[(size_t)k * E_SZ + e];
    pn.h[1] = (__bf16)w_noise[(size_t)(k + 1) * E_SZ + e];
    *(unsigned*)&wt_r[e * WSTRIDE + k] = pr.u;
    *(unsigned*)&wt_n[e * WSTRIDE + k] = pn.u;
  }
  __syncthreads();

  const int lane = threadIdx.x & 31;
  const int wave = threadIdx.x >> 5;
  const int tile = blockIdx.x * (blockDim.x >> 5) + wave;
  if (tile >= NTILES) return;           // uniform per-wave (grid covers exactly)

  const int half = lane >> 4;           // 0: lanes 0-15, 1: lanes 16-31
  const int ln16 = lane & 15;

  // A-fragment feeder row for this lane (A: lanes 0-15 and 16-31 both M=0..15)
  const float* xrow = x + (size_t)(tile * 16 + ln16) * D_SZ;

  v8f c_r = {};   // route accumulator  (C: 16x16 f32)
  v8f c_n = {};   // noise accumulator

  for (int k0 = 0; k0 < D_SZ; k0 += 32) {
    // Speculative prefetch of a future x chunk (OOB translation drops silently)
    __builtin_prefetch(xrow + k0 + 512, 0, 1);

    // ---- A fragment: 16x32 bf16 ----
    // lanes<16: VGPR0-3 hold K=k0+0..7, VGPR4-7 hold K=k0+16..23
    // lanes>=16: VGPR0-3 hold K=k0+8..15, VGPR4-7 hold K=k0+24..31
    const float* pa0 = xrow + k0 + half * 8;
    const float* pa1 = xrow + k0 + 16 + half * 8;
    f32x4 a0 = *(const f32x4*)(pa0);
    f32x4 a1 = *(const f32x4*)(pa0 + 4);
    f32x4 a2 = *(const f32x4*)(pa1);
    f32x4 a3 = *(const f32x4*)(pa1 + 4);
    v16bf a;
    a[0]  = (__bf16)a0.x; a[1]  = (__bf16)a0.y; a[2]  = (__bf16)a0.z; a[3]  = (__bf16)a0.w;
    a[4]  = (__bf16)a1.x; a[5]  = (__bf16)a1.y; a[6]  = (__bf16)a1.z; a[7]  = (__bf16)a1.w;
    a[8]  = (__bf16)a2.x; a[9]  = (__bf16)a2.y; a[10] = (__bf16)a2.z; a[11] = (__bf16)a2.w;
    a[12] = (__bf16)a3.x; a[13] = (__bf16)a3.y; a[14] = (__bf16)a3.z; a[15] = (__bf16)a3.w;

    // ---- B fragments from LDS (already bf16, transposed): lane = column N ----
    // lanes<16 need K=k0+0..7,16..23, lanes>=16 need K=k0+8..15,24..31 (matches A)
    const int kb = k0 + half * 8;
    const __bf16* pr = wt_r + ln16 * WSTRIDE + kb;
    const __bf16* pn = wt_n + ln16 * WSTRIDE + kb;
    v8bf rlo = *(const v8bf*)pr;
    v8bf rhi = *(const v8bf*)(pr + 16);
    v8bf nlo = *(const v8bf*)pn;
    v8bf nhi = *(const v8bf*)(pn + 16);
    v16bf br = __builtin_shufflevector(rlo, rhi, 0,1,2,3,4,5,6,7,8,9,10,11,12,13,14,15);
    v16bf bn = __builtin_shufflevector(nlo, nhi, 0,1,2,3,4,5,6,7,8,9,10,11,12,13,14,15);

    // Shared A fragment drives both accumulators
    c_r = __builtin_amdgcn_wmma_f32_16x16x32_bf16(false, a, false, br,
                                                  (short)0, c_r, false, false);
    c_n = __builtin_amdgcn_wmma_f32_16x16x32_bf16(false, a, false, bn,
                                                  (short)0, c_n, false, false);
  }

  // ---- Epilogue: bias, noisy logits, top-2, sparse softmax ----
  const float bias_r = b_route[ln16];
  const float bias_n = b_noise[ln16];

  float acc = 0.0f;   // per-(lane=expert) partial prob sum over this tile's rows
#pragma unroll
  for (int r = 0; r < 8; ++r) {
    const int row = tile * 16 + half * 8 + r;   // C layout: M = r + 8*half
    float logit = c_r[r] + bias_r;
    float nz    = c_n[r] + bias_n;
    // branch-free stable softplus with HW transcendentals
    float sp = fmaxf(nz, 0.0f) + __logf(1.0f + __expf(-fabsf(nz)));
    float ev = eps[(size_t)row * E_SZ + ln16];
    float z  = logit + ev * sp;                 // TEMPERATURE == 1

    // top-2 over the 16 experts held across lanes [half*16 .. half*16+15]
    float v1 = z, v2 = -INFINITY;
    int   i1 = ln16, i2 = -1;
#pragma unroll
    for (int m = 1; m < 16; m <<= 1) {
      float ov1 = __shfl_xor(v1, m, 32);
      float ov2 = __shfl_xor(v2, m, 32);
      int   oi1 = __shfl_xor(i1, m, 32);
      int   oi2 = __shfl_xor(i2, m, 32);
      float nv1, nv2; int ni1, ni2;
      if (ov1 > v1) {
        nv1 = ov1; ni1 = oi1;
        nv2 = (v1 > ov2) ? v1 : ov2;
        ni2 = (v1 > ov2) ? i1 : oi2;
      } else {
        nv1 = v1;  ni1 = i1;
        nv2 = (ov1 > v2) ? ov1 : v2;
        ni2 = (ov1 > v2) ? oi1 : i2;
      }
      v1 = nv1; v2 = nv2; i1 = ni1; i2 = ni2;
    }

    // softmax over {v1, v2}; everything else is exactly zero
    float w2 = __expf(v2 - v1);
    float invZ = 1.0f / (1.0f + w2);
    float p = 0.0f;
    if (ln16 == i1)      p = invZ;
    else if (ln16 == i2) p = w2 * invZ;

    out_probs[(size_t)row * E_SZ + ln16] = p;
    if (ln16 == 0) {
      out_idx[row * 2 + 0] = i1;
      out_idx[row * 2 + 1] = i2;
    }
    acc += p;
  }

  // fold the two lane-halves (same expert, different row groups), then atomics
  acc += __shfl_xor(acc, 16, 32);
  if (half == 0) atomicAdd(&aux_part[ln16], acc);
}

__global__ void aux_loss_finalize_kernel(const float* __restrict__ aux_part,
                                         float* __restrict__ out_aux)
{
  if (threadIdx.x == 0) {
    const float inv_n = 1.0f / (float)ROWS;
    const float target = 1.0f / (float)E_SZ;
    float s = 0.0f;
#pragma unroll
    for (int e = 0; e < E_SZ; ++e) {
      float d = aux_part[e] * inv_n - target;
      s += d * d;
    }
    *out_aux = s;
  }
}

extern "C" void kernel_launch(void* const* d_in, const int* in_sizes, int n_in,
                              void* d_out, int out_size, void* d_ws, size_t ws_size,
                              hipStream_t stream) {
  const float* x       = (const float*)d_in[0];
  const float* w_route = (const float*)d_in[1];
  const float* b_route = (const float*)d_in[2];
  const float* w_noise = (const float*)d_in[3];
  const float* b_noise = (const float*)d_in[4];
  const float* eps     = (const float*)d_in[5];

  float* out_probs = (float*)d_out;                           // ROWS*E floats
  int*   out_idx   = (int*)(out_probs + (size_t)ROWS * E_SZ); // ROWS*2 int32
  float* out_aux   = out_probs + (size_t)ROWS * E_SZ + (size_t)ROWS * K_TOP;

  float* aux_part = (float*)d_ws;                             // 16 floats
  hipMemsetAsync(aux_part, 0, E_SZ * sizeof(float), stream);

  const int waves_per_block = 8;                              // 256 threads
  const int blocks = NTILES / waves_per_block;                // 128
  const size_t smem_bytes = (size_t)2 * E_SZ * WSTRIDE * sizeof(__bf16); // 131,584 B
  noisy_topk_router_kernel<<<blocks, waves_per_block * 32, smem_bytes, stream>>>(
      x, w_route, b_route, w_noise, b_noise, eps,
      out_probs, out_idx, aux_part);

  aux_loss_finalize_kernel<<<1, 32, 0, stream>>>(aux_part, out_aux);
}
